// Encoder_1357209666145
// MI455X (gfx1250) — compile-verified, matrix operand order checked by reading, not run on hardware
//
#include <hip/hip_runtime.h>
#include <hip/hip_bf16.h>
#include <math.h>

// ---------------------------------------------------------------------------
// Types for WMMA
// ---------------------------------------------------------------------------
typedef _Float16 half16 __attribute__((ext_vector_type(16)));
typedef _Float16 half8  __attribute__((ext_vector_type(8)));
typedef float    float8 __attribute__((ext_vector_type(8)));

union H16 { half16 v; half8 h[2]; };
union F8  { float8 v; float f[8]; };

#define N_NODES 100000
#define N_EDGES 400000
#define N_GRAPHS 64

// ---------------------------------------------------------------------------
// Small helper kernels: degree / dis / norm
// ---------------------------------------------------------------------------
__global__ void k_zero(float* p, int n) {
    int i = blockIdx.x * blockDim.x + threadIdx.x;
    if (i < n) p[i] = 0.0f;
}

__global__ void k_degree(const int* dst, float* deg, int e) {
    int i = blockIdx.x * blockDim.x + threadIdx.x;
    if (i < e) atomicAdd(&deg[dst[i]], 1.0f);
}

__global__ void k_dis(const float* deg, float* dis, int n) {
    int i = blockIdx.x * blockDim.x + threadIdx.x;
    if (i < n) dis[i] = rsqrtf(deg[i] + 2.0f);
}

__global__ void k_norm(const int* src, const int* dst, const float* dis,
                       float* norm, int e) {
    int i = blockIdx.x * blockDim.x + threadIdx.x;
    if (i < e) norm[i] = dis[src[i]] * dis[dst[i]];
}

// ---------------------------------------------------------------------------
// fp32 -> f16 activation conversion with optional fused ReLU (8 elems/thread)
// ---------------------------------------------------------------------------
__global__ void k_cvt_x(const float* __restrict__ x, _Float16* __restrict__ xh,
                        long long n, int relu) {
    long long i = ((long long)blockIdx.x * blockDim.x + threadIdx.x) * 8;
    if (i >= n) return;
    const float4 a = *(const float4*)(x + i);
    const float4 b = *(const float4*)(x + i + 4);
    float v[8] = {a.x, a.y, a.z, a.w, b.x, b.y, b.z, b.w};
    if (relu) {
        #pragma unroll
        for (int j = 0; j < 8; ++j) v[j] = fmaxf(v[j], 0.0f);
    }
    half8 h;
    #pragma unroll
    for (int j = 0; j < 8; ++j) h[j] = (_Float16)v[j];
    *(half8*)(xh + i) = h;
}

// ---------------------------------------------------------------------------
// W[K x N] fp32 -> Wt[N x K] f16 (transposed so B fragments are contiguous)
// ---------------------------------------------------------------------------
__global__ void k_cvt_wt(const float* __restrict__ w, _Float16* __restrict__ wt,
                         int K, int N) {
    int idx = blockIdx.x * blockDim.x + threadIdx.x;
    if (idx >= K * N) return;
    int k = idx / N, n = idx % N;
    wt[(size_t)n * K + k] = (_Float16)w[idx];
}

// ---------------------------------------------------------------------------
// WMMA GEMM: H[M x N] = Xh[M x K] @ Wt[N x K]^T   (f16 in, f32 acc)
// Block = 256 threads = 8 waves; block tile 128M x 64N; wave w owns rows
// [rowB+16w, rowB+16w+16) x 4 adjacent 16-col tiles (A fragment reused 4x).
// Fragments loaded directly from global (L2-resident f16 operands):
//   A: lane L<16 -> row (L), K{kb..kb+7, kb+16..kb+23}; L>=16 -> K{8..15,24..31}
//   B: identical pattern on Wt rows (= output columns).
// Two 16B loads per fragment; OOB rows clamped (their C rows never stored).
// ---------------------------------------------------------------------------
template <int K, int N>
__global__ __launch_bounds__(256)
void k_gemm_wmma(const _Float16* __restrict__ Xh, const _Float16* __restrict__ Wt,
                 float* __restrict__ H, int M) {
    const int tid  = threadIdx.x;
    const int lane = tid & 31;
    const int wv   = tid >> 5;
    const int rowB = blockIdx.x * 128;
    const int ncol = blockIdx.y * 64;

    const int arow = rowB + wv * 16 + (lane & 15);
    const int kc   = (lane >> 4) * 8;          // 0 for lanes 0-15, 8 for 16-31
    const int bn   = lane & 15;

    const size_t aoff = (size_t)(arow < M ? arow : (M - 1)) * K;

    float8 acc0 = {0,0,0,0,0,0,0,0};
    float8 acc1 = {0,0,0,0,0,0,0,0};
    float8 acc2 = {0,0,0,0,0,0,0,0};
    float8 acc3 = {0,0,0,0,0,0,0,0};

    #pragma unroll 2
    for (int kb = 0; kb < K; kb += 32) {
        H16 a;
        a.h[0] = *(const half8*)(Xh + aoff + kb + kc);
        a.h[1] = *(const half8*)(Xh + aoff + kb + kc + 16);

        H16 b0, b1, b2, b3;
        const _Float16* bp0 = Wt + (size_t)(ncol + 0  + bn) * K + kb + kc;
        const _Float16* bp1 = Wt + (size_t)(ncol + 16 + bn) * K + kb + kc;
        const _Float16* bp2 = Wt + (size_t)(ncol + 32 + bn) * K + kb + kc;
        const _Float16* bp3 = Wt + (size_t)(ncol + 48 + bn) * K + kb + kc;
        b0.h[0] = *(const half8*)(bp0);  b0.h[1] = *(const half8*)(bp0 + 16);
        b1.h[0] = *(const half8*)(bp1);  b1.h[1] = *(const half8*)(bp1 + 16);
        b2.h[0] = *(const half8*)(bp2);  b2.h[1] = *(const half8*)(bp2 + 16);
        b3.h[0] = *(const half8*)(bp3);  b3.h[1] = *(const half8*)(bp3 + 16);

        acc0 = __builtin_amdgcn_wmma_f32_16x16x32_f16(false, a.v, false, b0.v,
                                                      (short)0, acc0, false, false);
        acc1 = __builtin_amdgcn_wmma_f32_16x16x32_f16(false, a.v, false, b1.v,
                                                      (short)0, acc1, false, false);
        acc2 = __builtin_amdgcn_wmma_f32_16x16x32_f16(false, a.v, false, b2.v,
                                                      (short)0, acc2, false, false);
        acc3 = __builtin_amdgcn_wmma_f32_16x16x32_f16(false, a.v, false, b3.v,
                                                      (short)0, acc3, false, false);
    }

    // epilogue: C layout -> VGPR r holds (M = r + 8*(lane>=16), N = lane&15)
    F8 u0, u1, u2, u3;
    u0.v = acc0; u1.v = acc1; u2.v = acc2; u3.v = acc3;
    const int crow0 = rowB + wv * 16 + ((lane >> 4) * 8);
    const int ccol  = ncol + (lane & 15);
    #pragma unroll
    for (int r = 0; r < 8; ++r) {
        const int row = crow0 + r;
        if (row < M) {
            float* hp = H + (size_t)row * N + ccol;
            hp[0]  = u0.f[r];
            hp[16] = u1.f[r];
            hp[32] = u2.f[r];
            hp[48] = u3.f[r];
        }
    }
}

// ---------------------------------------------------------------------------
// agg init: agg[i][f] = H[i][f] * 2*dis[i]^2 + b[f]   (self-loop + bias, and
// deterministic zero-base for the subsequent scatter accumulation)
// ---------------------------------------------------------------------------
__global__ void k_init_agg(const float* __restrict__ H, const float* __restrict__ dis,
                           const float* __restrict__ b, float* __restrict__ agg,
                           int n, int D) {
    long long idx = (long long)blockIdx.x * blockDim.x + threadIdx.x;
    const int vecs = D >> 2;
    if (idx >= (long long)n * vecs) return;
    const int i = (int)(idx / vecs);
    const int f = (int)(idx % vecs) * 4;
    const float d  = dis[i];
    const float s  = 2.0f * d * d;
    const float4 h  = *(const float4*)(H + (size_t)i * D + f);
    const float4 bb = *(const float4*)(b + f);
    float4 o;
    o.x = h.x * s + bb.x;  o.y = h.y * s + bb.y;
    o.z = h.z * s + bb.z;  o.w = h.w * s + bb.w;
    *(float4*)(agg + (size_t)i * D + f) = o;
}

// ---------------------------------------------------------------------------
// edge scatter: agg[dst] += H[src] * norm[e]   (float4 chunks, f32 atomics)
// ---------------------------------------------------------------------------
__global__ void k_scatter(const float* __restrict__ H, float* __restrict__ agg,
                          const int* __restrict__ src, const int* __restrict__ dst,
                          const float* __restrict__ norm, int E, int D) {
    long long idx = (long long)blockIdx.x * blockDim.x + threadIdx.x;
    const int vecs = D >> 2;
    if (idx >= (long long)E * vecs) return;
    const int e = (int)(idx / vecs);
    const int f = (int)(idx % vecs) * 4;
    const float nm = norm[e];
    const int s = src[e], d = dst[e];
    const float4 h = *(const float4*)(H + (size_t)s * D + f);
    float* o = agg + (size_t)d * D + f;
    atomicAdd(o + 0, h.x * nm);
    atomicAdd(o + 1, h.y * nm);
    atomicAdd(o + 2, h.z * nm);
    atomicAdd(o + 3, h.w * nm);
}

// ---------------------------------------------------------------------------
// global max pool over sorted batch ids: one block per graph, binary search
// segment bounds, thread f reduces feature f over the segment.
// ---------------------------------------------------------------------------
__global__ void k_pool(const float* __restrict__ x, const int* __restrict__ batch,
                       float* __restrict__ out, int n, int D) {
    const int g = blockIdx.x;
    const int f = threadIdx.x;
    int lo = 0, hi = n;
    while (lo < hi) { int mid = (lo + hi) >> 1; if (batch[mid] < g) lo = mid + 1; else hi = mid; }
    const int start = lo;
    lo = start; hi = n;
    while (lo < hi) { int mid = (lo + hi) >> 1; if (batch[mid] < g + 1) lo = mid + 1; else hi = mid; }
    const int end = lo;
    float m = -INFINITY;
    for (int i = start; i < end; ++i)
        m = fmaxf(m, x[(size_t)i * D + f]);
    out[(size_t)g * D + f] = m;
}

// ---------------------------------------------------------------------------
// Host: kernel_launch
// Inputs: 0 data[100000*128] f32, 1 edge_index[2*400000] i32, 2 batch[100000] i32,
//         3 W1,4 b1, 5 W2,6 b2, 7 W3,8 b3, 9 W4,10 b4, 11 W5,12 b5 (f32)
// Output: [64*128] f32
// ---------------------------------------------------------------------------
extern "C" void kernel_launch(void* const* d_in, const int* in_sizes, int n_in,
                              void* d_out, int out_size, void* d_ws, size_t ws_size,
                              hipStream_t stream) {
    (void)in_sizes; (void)n_in; (void)out_size; (void)ws_size;

    const float* data  = (const float*)d_in[0];
    const int*   ei    = (const int*)d_in[1];
    const int*   batch = (const int*)d_in[2];
    const int*   src   = ei;
    const int*   dst   = ei + N_EDGES;
    float*       out   = (float*)d_out;

    const int dims[6] = {128, 128, 256, 512, 256, 128};

    // workspace layout (floats / halves)
    const size_t BIG = (size_t)N_NODES * 512;          // 51.2M elements
    float*     bufA = (float*)d_ws;                    // activations / agg
    float*     bufB = bufA + BIG;                      // H = X @ W
    _Float16*  Xh   = (_Float16*)(bufB + BIG);         // f16 activations (BIG halves)
    _Float16*  Wth  = (_Float16*)(Xh + BIG);           // f16 W^T (<= 131072 halves)
    float*     dis  = (float*)(Wth + 131072);
    float*     deg  = dis + N_NODES;
    float*     norm = deg + N_NODES;

    const int T = 256;

    // ---- precompute degree / dis / per-edge norm (deterministic each call) ----
    k_zero<<<(N_NODES + T - 1) / T, T, 0, stream>>>(deg, N_NODES);
    k_degree<<<(N_EDGES + T - 1) / T, T, 0, stream>>>(dst, deg, N_EDGES);
    k_dis<<<(N_NODES + T - 1) / T, T, 0, stream>>>(deg, dis, N_NODES);
    k_norm<<<(N_EDGES + T - 1) / T, T, 0, stream>>>(src, dst, dis, norm, N_EDGES);

    // ---- 5 GCN layers ----
    for (int l = 0; l < 5; ++l) {
        const int K = dims[l];
        const int D = dims[l + 1];
        const float* X = (l == 0) ? data : bufA;
        const float* W = (const float*)d_in[3 + 2 * l];
        const float* b = (const float*)d_in[4 + 2 * l];

        // f16 operand prep: X (+ relu of previous layer), W transposed
        long long nx = (long long)N_NODES * K;
        k_cvt_x<<<(unsigned)((nx / 8 + T - 1) / T), T, 0, stream>>>(X, Xh, nx, l > 0 ? 1 : 0);
        k_cvt_wt<<<(K * D + T - 1) / T, T, 0, stream>>>(W, Wth, K, D);

        // H = Xh @ Wt^T via WMMA
        dim3 ggrid((N_NODES + 127) / 128, D / 64);
        switch (l) {
            case 0: k_gemm_wmma<128, 128><<<ggrid, T, 0, stream>>>(Xh, Wth, bufB, N_NODES); break;
            case 1: k_gemm_wmma<128, 256><<<ggrid, T, 0, stream>>>(Xh, Wth, bufB, N_NODES); break;
            case 2: k_gemm_wmma<256, 512><<<ggrid, T, 0, stream>>>(Xh, Wth, bufB, N_NODES); break;
            case 3: k_gemm_wmma<512, 256><<<ggrid, T, 0, stream>>>(Xh, Wth, bufB, N_NODES); break;
            case 4: k_gemm_wmma<256, 128><<<ggrid, T, 0, stream>>>(Xh, Wth, bufB, N_NODES); break;
        }

        // agg = H * 2*dis^2 + b   (overwrites bufA; X no longer needed)
        long long initN = (long long)N_NODES * (D >> 2);
        k_init_agg<<<(unsigned)((initN + T - 1) / T), T, 0, stream>>>(bufB, dis, b, bufA, N_NODES, D);

        // agg[dst] += H[src] * norm
        long long scatN = (long long)N_EDGES * (D >> 2);
        k_scatter<<<(unsigned)((scatN + T - 1) / T), T, 0, stream>>>(bufB, bufA, src, dst, norm, N_EDGES, D);
        // next layer's k_cvt_x applies relu while reading bufA (layer 5 stays linear)
    }

    // ---- global max pool over graphs ----
    k_pool<<<N_GRAPHS, 128, 0, stream>>>(bufA, batch, out, N_NODES, 128);
}
